// MGU_20804821582086
// MI455X (gfx1250) — matrix operand order repaired and checked by reading, not compiled, not run on hardware
//
#include <hip/hip_runtime.h>
#include <cstdint>

// MGU on MI455X (gfx1250, wave32).
// Phase 1: Xp = x @ W_ih^T + b_ih as one parallel bf16-WMMA GEMM, written into d_out.
// Phase 2: 32 persistent WGs run the 1024-step recurrence; W_hh slice cached in LDS
//          as bf16, h exchanged via ping-pong bf16 buffer + global-atomic grid barrier.
// bf16 packing is a single v_perm_b32 (truncate); WMMA accumulates in f32.

typedef __attribute__((ext_vector_type(16))) __bf16 v16bf;
typedef __attribute__((ext_vector_type(8)))  float  v8f;

#define kT 1024
#define kB 64
#define kF 256
#define kH 512

union FragU { uint4 u[2]; v16bf v; };

__device__ inline unsigned fbits(float x) {
  union { float f; unsigned u; } v; v.f = x; return v.u;
}
// pack {lo=bf16(a), hi=bf16(b)} in ONE v_perm_b32:
// dst bytes [1:0] = a bytes [3:2] (sel 6,7), dst bytes [3:2] = b bytes [3:2] (sel 2,3)
__device__ inline unsigned pk2(float a, float b) {
  return __builtin_amdgcn_perm(fbits(a), fbits(b), 0x03020706u);
}
__device__ inline unsigned short bf16c(float x) {   // truncate; folds to d16_hi store
  return (unsigned short)(fbits(x) >> 16);
}

// ---------------- Phase 1: Xp[65536,512] = x[65536,256] @ W_ih^T + b_ih ----------------
// WG = 256 thr (8 waves). 128x128 output tile, K staged 32 at a time through LDS as bf16.
// Wave (wm,wn) computes a 32x64 sub-tile = 2x4 WMMA accumulators.
__global__ __launch_bounds__(256) void mgu_xproj(
    const float* __restrict__ x, const float* __restrict__ Wih,
    const float* __restrict__ bih, float* __restrict__ Xp) {
  __shared__ unsigned As[128 * 16];  // 128 rows x 32 bf16 (row-major)
  __shared__ unsigned Bs[128 * 16];  // 128 cols x 32 bf16 (col-of-B = row-of-Wih, contiguous K)
  const int tid  = threadIdx.x;
  const int lane = tid & 31, wave = tid >> 5;
  const int g = lane >> 4, lm = lane & 15;
  const int wm = wave >> 1, wn = wave & 1;
  const int m0 = blockIdx.y * 128, n0 = blockIdx.x * 128;

  v8f acc[2][4];
#pragma unroll
  for (int a = 0; a < 2; ++a)
#pragma unroll
    for (int b = 0; b < 4; ++b)
#pragma unroll
      for (int i = 0; i < 8; ++i) acc[a][b][i] = 0.0f;

  const int r = tid >> 1, hq = tid & 1;                 // 2 threads per tile row
  const float* srcA = x   + (size_t)(m0 + r) * kF + hq * 16;
  const float* srcB = Wih + (size_t)(n0 + r) * kF + hq * 16;

  for (int kk = 0; kk < kF; kk += 32) {
    __syncthreads();
    {
      const float4* pa = (const float4*)(srcA + kk);
      const float4* pb = (const float4*)(srcB + kk);
      unsigned* da = &As[r * 16 + hq * 8];
      unsigned* db = &Bs[r * 16 + hq * 8];
#pragma unroll
      for (int q = 0; q < 4; ++q) {
        float4 fa = pa[q], fb = pb[q];
        da[q * 2]     = pk2(fa.x, fa.y);
        da[q * 2 + 1] = pk2(fa.z, fa.w);
        db[q * 2]     = pk2(fb.x, fb.y);
        db[q * 2 + 1] = pk2(fb.z, fb.w);
      }
      if (kk + 32 < kF) {   // gfx1250 global_prefetch_b8; locality 3 -> WGP-scope fill
        __builtin_prefetch(srcA + kk + 32, 0, 3);
        __builtin_prefetch(srcB + kk + 32, 0, 3);
      }
    }
    __syncthreads();

    FragU a[2], b[4];
#pragma unroll
    for (int dm = 0; dm < 2; ++dm) {      // A frag: two 8-bf16 K-runs per lane
      int row = wm * 32 + dm * 16 + lm;
      a[dm].u[0] = *(const uint4*)&As[row * 16 + g * 4];
      a[dm].u[1] = *(const uint4*)&As[row * 16 + 8 + g * 4];
    }
#pragma unroll
    for (int dn = 0; dn < 4; ++dn) {      // B frag: two 8-bf16 K-runs per lane
      int col = wn * 64 + dn * 16 + lm;
      b[dn].u[0] = *(const uint4*)&Bs[col * 16 + g * 4];
      b[dn].u[1] = *(const uint4*)&Bs[col * 16 + 8 + g * 4];
    }
#pragma unroll
    for (int dm = 0; dm < 2; ++dm)
#pragma unroll
      for (int dn = 0; dn < 4; ++dn)
        acc[dm][dn] = __builtin_amdgcn_wmma_f32_16x16x32_bf16(
            false, a[dm].v, false, b[dn].v, (short)0, acc[dm][dn], false, false);
  }

#pragma unroll
  for (int dm = 0; dm < 2; ++dm)
#pragma unroll
    for (int dn = 0; dn < 4; ++dn) {
      int cg = n0 + wn * 64 + dn * 16 + lm;
      float bias = bih[cg];
      int rbase = m0 + wm * 32 + dm * 16 + g * 8;  // C layout: VGPR i -> M = base+g*8+i
#pragma unroll
      for (int i = 0; i < 8; ++i)
        Xp[(size_t)(rbase + i) * kH + cg] = acc[dm][dn][i] + bias;
    }
}

// ---------------- Phase 2: persistent recurrence ----------------
// 32 WGs x 128 thr (4 waves). WG wg owns h-columns [wg*16, wg*16+16).
// LDS: W_hh rows {c0..c0+15, 512+c0..512+c0+15} as bf16 (32 KB) + fp32 h slice (4 KB).
// Wave w = M-tile (batch rows w*16..w*16+15); computes BOTH gate tiles (f and n).
__global__ __launch_bounds__(128) void mgu_recur(
    const float* __restrict__ Whh, const float* __restrict__ bhh,
    unsigned short* __restrict__ hbf, float* __restrict__ out,
    unsigned* __restrict__ bar, int nwg) {
  __shared__ unsigned Ws[32 * 256];   // 32 rows x 512 bf16
  __shared__ float hloc[64 * 16];     // fp32 master copy of this WG's h columns
  const int tid  = threadIdx.x;
  const int lane = tid & 31, wave = tid >> 5;
  const int g = lane >> 4, lm = lane & 15;
  const int c0 = blockIdx.x * 16;
  const int mbase = wave * 16;

  // Preload W_hh slice -> LDS bf16 (once for all 1024 steps).
  for (int it = tid; it < 4096; it += 128) {         // 32 rows * 128 float4
    int s = it >> 7, q = it & 127;
    int grow = (s < 16) ? (c0 + s) : (kH + c0 + (s - 16));
    float4 f = ((const float4*)(Whh + (size_t)grow * kH))[q];
    Ws[s * 256 + q * 2]     = pk2(f.x, f.y);
    Ws[s * 256 + q * 2 + 1] = pk2(f.z, f.w);
  }
  for (int i = tid; i < 64 * 16; i += 128) hloc[i] = 0.0f;  // h0 = 0
  __syncthreads();

  const float bfv = bhh[c0 + lm];
  const float bnv = bhh[kH + c0 + lm];

  for (int t = 0; t < kT; ++t) {
    const unsigned short* hr = hbf + (size_t)(t & 1) * (kB * kH);
    unsigned short*       hw = hbf + (size_t)((t + 1) & 1) * (kB * kH);
    v8f accF, accN;
#pragma unroll
    for (int i = 0; i < 8; ++i) { accF[i] = 0.0f; accN[i] = 0.0f; }

    const unsigned short* arow = hr + (size_t)(mbase + lm) * kH;
#pragma unroll 4
    for (int kkk = 0; kkk < 16; ++kkk) {              // K = 512 in steps of 32
      FragU A, Bf, Bn;
      A.u[0] = *(const uint4*)(arow + kkk * 32 + g * 16);
      A.u[1] = *(const uint4*)(arow + kkk * 32 + g * 16 + 8);
      int bi  = lm * 256        + kkk * 16 + g * 8;   // forget-gate W row
      int bi2 = (16 + lm) * 256 + kkk * 16 + g * 8;   // new-gate W row
      Bf.u[0] = *(const uint4*)&Ws[bi];   Bf.u[1] = *(const uint4*)&Ws[bi + 4];
      Bn.u[0] = *(const uint4*)&Ws[bi2];  Bn.u[1] = *(const uint4*)&Ws[bi2 + 4];
      accF = __builtin_amdgcn_wmma_f32_16x16x32_bf16(false, A.v, false, Bf.v,
                                                     (short)0, accF, false, false);
      accN = __builtin_amdgcn_wmma_f32_16x16x32_bf16(false, A.v, false, Bn.v,
                                                     (short)0, accN, false, false);
    }

    const size_t tb = (size_t)t * kB;
    const int cg = c0 + lm;
#pragma unroll
    for (int i = 0; i < 8; ++i) {
      int row = mbase + g * 8 + i;
      float fgate = 1.0f / (1.0f + __expf(-(accF[i] + bfv)));
      float hn = accN[i] + bnv;
      size_t oidx = (tb + row) * (size_t)kH + cg;
      float xin = out[oidx];                       // i_n written by phase 1; same lane,
      float n = tanhf(xin + fgate * hn);           // read-once then overwritten below
      float hold = hloc[row * 16 + lm];
      float hnew = n + (1.0f - fgate) * (hold - n);
      hloc[row * 16 + lm] = hnew;
      out[oidx] = hnew;
      hw[(size_t)row * kH + cg] = bf16c(hnew);     // publish bf16 h for next step
    }

    // grid-wide step barrier: monotonic counter, zeroed per launch (deterministic)
    __threadfence();
    __syncthreads();
    if (tid == 0) {
      __hip_atomic_fetch_add(bar, 1u, __ATOMIC_RELEASE, __HIP_MEMORY_SCOPE_AGENT);
      unsigned target = (unsigned)(t + 1) * (unsigned)nwg;
      while (__hip_atomic_load(bar, __ATOMIC_ACQUIRE, __HIP_MEMORY_SCOPE_AGENT) < target)
        __builtin_amdgcn_s_sleep(8);
    }
    __syncthreads();
    __threadfence();
  }
}

extern "C" void kernel_launch(void* const* d_in, const int* in_sizes, int n_in,
                              void* d_out, int out_size, void* d_ws, size_t ws_size,
                              hipStream_t stream) {
  const float* x   = (const float*)d_in[0];
  const float* Wih = (const float*)d_in[1];
  const float* Whh = (const float*)d_in[2];
  const float* bih = (const float*)d_in[3];
  const float* bhh = (const float*)d_in[4];
  float* out = (float*)d_out;

  // ws: [2 x 64x512 bf16 h ping-pong][barrier counter]
  unsigned short* hbf = (unsigned short*)d_ws;
  unsigned* bar = (unsigned*)((char*)d_ws + (size_t)2 * kB * kH * sizeof(unsigned short));
  hipMemsetAsync(d_ws, 0, (size_t)2 * kB * kH * sizeof(unsigned short) + 64, stream);

  // Phase 1: 65536x512 GEMM, 128x128 tiles -> grid (4, 512)
  dim3 gA(kH / 128, (kT * kB) / 128);
  mgu_xproj<<<gA, 256, 0, stream>>>(x, Wih, bih, out);

  // Phase 2: 32 persistent workgroups
  const int nwg = kH / 16;
  mgu_recur<<<dim3(nwg), 128, 0, stream>>>(Whh, bhh, hbf, out, bar, nwg);
}
